// PairwiseMaxMarginHingeLossFunc_78082505441520
// MI455X (gfx1250) — matrix operand order repaired and checked by reading, not compiled, not run on hardware
//
#include <hip/hip_runtime.h>
#include <hip/hip_bf16.h>

// ---------------------------------------------------------------------------
// Pairwise max-margin hinge loss + argmax accuracy, CDNA5 (gfx1250, wave32).
// N=2048, D=128, F=8.
//  sim[f] = normalize_rows(Q_f) @ normalize_rows(D_f)^T  via v_wmma_f32_16x16x32_bf16
//  loss   = sum max(margin - (sim_diag - sim)*(lab_diag - lab), 0) off-diagonal
//           (computed branchlessly over ALL elements; each diagonal element
//            contributes exactly max(margin,0) since lab_diag-lab == ±0 there,
//            so F*N*max(margin,0) is subtracted once at the end)
//  count  = sum over (f,row) of [argmax_j sim == argmax_j label]
// ---------------------------------------------------------------------------

typedef __attribute__((ext_vector_type(16))) __bf16 v16bf;
typedef __attribute__((ext_vector_type(8)))  float  v8f;

#define NN   2048
#define DD   128
#define FF   8
#define NT   (NN / 16)          // 128 row/col tiles
#define KT   (DD / 32)          // 4 k tiles of 32
#define TILE_ELEMS 512          // 32 lanes * 16 bf16 per packed WMMA tile
#define NEG_BIG (-1.0e30f)

// ---------------------------------------------------------------------------
// Kernel 1: per (f,n) wave — fp32 norms, fp32 diagonal cosine, bf16 pack into
// the exact CDNA5 WMMA A(16x32) / B(32x16) bf16 VGPR layouts.
// A layout  (lane<16: m=lane, e<8 -> k=e, e>=8 -> k=e+8;
//            lane>=16: m=lane-16, e<8 -> k=e+8, e>=8 -> k=e+16)
// B layout  (same k order as A: lane<16: n=lane, e<8 -> k=e, e>=8 -> k=e+8;
//            lane>=16: n=lane-16, e<8 -> k=e+8, e>=8 -> k=e+16)
// ---------------------------------------------------------------------------
__global__ void cosim_prep_pack(const float* __restrict__ out_matrix,
                                __bf16* __restrict__ Apack,
                                __bf16* __restrict__ Bpack,
                                float* __restrict__ diag_s) {
    int gwave = (blockIdx.x * blockDim.x + threadIdx.x) >> 5;
    int lane  = threadIdx.x & 31;
    int f = gwave >> 11;          // / 2048
    int n = gwave & (NN - 1);
    if (f >= FF) return;

    // element (s,n,d,f) at ((s*N+n)*D+d)*F+f
    size_t base_q = ((size_t)n * DD) * FF + f;
    size_t base_d = (((size_t)NN + n) * DD) * FF + f;

    float q[4], dv[4];
    float sq = 0.f, sd = 0.f;
    #pragma unroll
    for (int t = 0; t < 4; ++t) {
        int d = lane + 32 * t;
        q[t]  = out_matrix[base_q + (size_t)d * FF];
        dv[t] = out_matrix[base_d + (size_t)d * FF];
        sq += q[t] * q[t];
        sd += dv[t] * dv[t];
    }
    #pragma unroll
    for (int mk = 16; mk >= 1; mk >>= 1) {
        sq += __shfl_xor(sq, mk, 32);
        sd += __shfl_xor(sd, mk, 32);
    }
    float invq = sq > 0.f ? rsqrtf(sq) : 0.f;
    float invd = sd > 0.f ? rsqrtf(sd) : 0.f;

    float dot = 0.f;
    #pragma unroll
    for (int t = 0; t < 4; ++t) {
        q[t]  *= invq;
        dv[t] *= invd;
        dot   += q[t] * dv[t];
    }
    #pragma unroll
    for (int mk = 16; mk >= 1; mk >>= 1) dot += __shfl_xor(dot, mk, 32);
    if (lane == 0) diag_s[f * NN + n] = dot;   // fp32 cosine diagonal

    int i_tile = n >> 4;
    int mrow   = n & 15;
    size_t tile_row = ((size_t)f * NT + i_tile) * KT;   // tile index base

    #pragma unroll
    for (int t = 0; t < 4; ++t) {
        int k  = lane + 32 * t;
        int kt = k >> 5;
        int kl = k & 31;
        size_t tbase = (tile_row + kt) * TILE_ELEMS;
        // A pack
        int al, ae;
        if (kl < 8)       { al = mrow;      ae = kl;      }
        else if (kl < 16) { al = mrow + 16; ae = kl - 8;  }
        else if (kl < 24) { al = mrow;      ae = kl - 8;  }
        else              { al = mrow + 16; ae = kl - 16; }
        Apack[tbase + al * 16 + ae] = (__bf16)q[t];
        // B pack
        int bl, be;
        if (kl < 8)       { bl = mrow;      be = kl;      }
        else if (kl < 16) { bl = mrow + 16; be = kl - 8;  }
        else if (kl < 24) { bl = mrow;      be = kl - 8;  }
        else              { bl = mrow + 16; be = kl - 16; }
        Bpack[tbase + bl * 16 + be] = (__bf16)dv[t];
    }
}

// ---------------------------------------------------------------------------
// Kernel 2: one workgroup (8 wave32) per (f, i_tile). Each wave sweeps
// interleaved j tiles: 4x WMMA (K=128), then a fully branchless fused
// hinge + argmax epilogue. Label matrix is read exactly once here
// (the 134 MB bandwidth floor).
// ---------------------------------------------------------------------------
__global__ void __launch_bounds__(256)
cosim_hinge_main(const __bf16* __restrict__ Apack,
                 const __bf16* __restrict__ Bpack,
                 const float* __restrict__ diag_s,
                 const float* __restrict__ label,
                 const float* __restrict__ margin_p,
                 float* __restrict__ partial_loss,
                 int*  __restrict__ partial_cnt) {
    int wg     = blockIdx.x;          // 0..1023
    int f      = wg / NT;
    int i_tile = wg % NT;
    int tid    = threadIdx.x;
    int wave   = tid >> 5;
    int lane   = tid & 31;
    int half   = lane >> 4;           // C layout: lane>=16 holds rows M=r+8
    int lcol   = lane & 15;
    int i0     = i_tile * 16;
    float margin = *margin_p;

    // A operand tiles, reused for all j (packed layout: contiguous 32B/lane)
    v16bf a[KT];
    size_t abase = ((size_t)f * NT + i_tile) * KT * TILE_ELEMS;
    #pragma unroll
    for (int kt = 0; kt < KT; ++kt)
        a[kt] = *(const v16bf*)(Apack + abase + (size_t)kt * TILE_ELEMS + lane * 16);

    // Per-lane diagonals for rows m = i0 + r + 8*half (r = C-accum VGPR index)
    float ds[8], dl[8];
    #pragma unroll
    for (int r = 0; r < 8; ++r) {
        int m = i0 + r + 8 * half;
        ds[r] = diag_s[f * NN + m];
        dl[r] = label[(size_t)f * NN * NN + (size_t)m * NN + m];
    }
    const float* lbase =
        label + (size_t)f * NN * NN + (size_t)(i0 + 8 * half) * NN + lcol;

    float lsum = 0.f;
    float smax[8], lmax[8];
    int   sidx[8], lidx[8];
    #pragma unroll
    for (int r = 0; r < 8; ++r) {
        smax[r] = NEG_BIG; lmax[r] = NEG_BIG; sidx[r] = 0; lidx[r] = 0;
    }

    for (int jt = wave; jt < NT; jt += 8) {
        int j0 = jt * 16;
        size_t bbase = ((size_t)f * NT + jt) * KT * TILE_ELEMS;
        v8f c = {};
        #pragma unroll
        for (int kt = 0; kt < KT; ++kt) {
            v16bf b = *(const v16bf*)(Bpack + bbase + (size_t)kt * TILE_ELEMS + lane * 16);
            c = __builtin_amdgcn_wmma_f32_16x16x32_bf16(
                    false, a[kt], false, b, (short)0, c, false, false);
        }
        // prefetch next j-tile's label row (speculative; OOB dropped)
        __builtin_prefetch(lbase + (size_t)(j0 + 128), 0, 1);

        int n = j0 + lcol;
        #pragma unroll
        for (int r = 0; r < 8; ++r) {
            float s   = c[r];
            float lab = lbase[(size_t)r * NN + j0];
            // argmax trackers (cndmask selects, no branches)
            bool cs = (s > smax[r]);
            smax[r] = cs ? s : smax[r];
            sidx[r] = cs ? n : sidx[r];
            bool cl = (lab > lmax[r]);
            lmax[r] = cl ? lab : lmax[r];
            lidx[r] = cl ? n : lidx[r];
            // branchless hinge: diagonal contributes exactly max(margin,0)
            // (dl[r]-lab == ±0 there), corrected globally in the finish kernel
            float h = fmaf(-(ds[r] - s), (dl[r] - lab), margin);
            lsum += fmaxf(h, 0.f);
        }
    }

    // Argmax reduce within each 16-lane half (rows differ across halves)
    #pragma unroll
    for (int r = 0; r < 8; ++r) {
        #pragma unroll
        for (int mk = 8; mk >= 1; mk >>= 1) {
            float ov = __shfl_xor(smax[r], mk, 32);
            int   oi = __shfl_xor(sidx[r], mk, 32);
            if (ov > smax[r] || (ov == smax[r] && oi < sidx[r])) { smax[r] = ov; sidx[r] = oi; }
            ov = __shfl_xor(lmax[r], mk, 32);
            oi = __shfl_xor(lidx[r], mk, 32);
            if (ov > lmax[r] || (ov == lmax[r] && oi < lidx[r])) { lmax[r] = ov; lidx[r] = oi; }
        }
    }
    #pragma unroll
    for (int mk = 16; mk >= 1; mk >>= 1) lsum += __shfl_xor(lsum, mk, 32);

    __shared__ float sh_sv[8][16]; __shared__ int sh_si[8][16];
    __shared__ float sh_lv[8][16]; __shared__ int sh_li[8][16];
    __shared__ float sh_loss[8];
    if (lcol == 0) {   // lanes 0 and 16: rows r and r+8
        #pragma unroll
        for (int r = 0; r < 8; ++r) {
            int row = r + 8 * half;
            sh_sv[wave][row] = smax[r]; sh_si[wave][row] = sidx[r];
            sh_lv[wave][row] = lmax[r]; sh_li[wave][row] = lidx[r];
        }
    }
    if (lane == 0) sh_loss[wave] = lsum;
    __syncthreads();

    if (tid < 16) {   // one thread per row, fixed wave order -> deterministic
        float bs = NEG_BIG, bl = NEG_BIG; int bsi = 0, bli = 0;
        for (int w = 0; w < 8; ++w) {
            float v = sh_sv[w][tid]; int i = sh_si[w][tid];
            if (v > bs || (v == bs && i < bsi)) { bs = v; bsi = i; }
            v = sh_lv[w][tid]; i = sh_li[w][tid];
            if (v > bl || (v == bl && i < bli)) { bl = v; bli = i; }
        }
        int cnt = (bsi == bli) ? 1 : 0;
        #pragma unroll
        for (int mk = 8; mk >= 1; mk >>= 1) cnt += __shfl_xor(cnt, mk, 32);
        if (tid == 0) {
            float tot = 0.f;
            for (int w = 0; w < 8; ++w) tot += sh_loss[w];
            partial_loss[wg] = tot;
            partial_cnt[wg]  = cnt;
        }
    }
}

// ---------------------------------------------------------------------------
// Kernel 3: deterministic fixed-order reduction of the 1024 partials,
// minus the F*N diagonal hinge terms (each exactly max(margin,0)).
// ---------------------------------------------------------------------------
__global__ void cosim_finish(const float* __restrict__ partial_loss,
                             const int*  __restrict__ partial_cnt,
                             const float* __restrict__ margin_p,
                             float* __restrict__ out) {
    __shared__ float sl[256];
    __shared__ int   sc[256];
    int t = threadIdx.x;
    float l = 0.f; int c = 0;
    for (int i = t; i < FF * NT; i += 256) { l += partial_loss[i]; c += partial_cnt[i]; }
    sl[t] = l; sc[t] = c;
    __syncthreads();
    for (int s = 128; s >= 1; s >>= 1) {
        if (t < s) { sl[t] += sl[t + s]; sc[t] += sc[t + s]; }
        __syncthreads();
    }
    if (t == 0) {
        float diag_corr = (float)(FF * NN) * fmaxf(*margin_p, 0.f);
        out[0] = sl[0] - diag_corr;
        out[1] = (float)sc[0];
    }
}

// ---------------------------------------------------------------------------
extern "C" void kernel_launch(void* const* d_in, const int* in_sizes, int n_in,
                              void* d_out, int out_size, void* d_ws, size_t ws_size,
                              hipStream_t stream) {
    const float* out_matrix = (const float*)d_in[0];   // [2,N,D,F] fp32
    const float* label      = (const float*)d_in[1];   // [F,N,N]   fp32
    const float* margin     = (const float*)d_in[2];   // scalar
    float* out = (float*)d_out;

    char* ws = (char*)d_ws;
    const size_t PACK_BYTES = (size_t)FF * NN * DD * sizeof(__bf16);  // 4 MiB each
    __bf16* Apack = (__bf16*)ws;
    __bf16* Bpack = (__bf16*)(ws + PACK_BYTES);
    float*  diag  = (float*)(ws + 2 * PACK_BYTES);
    float*  ploss = (float*)(ws + 2 * PACK_BYTES + (size_t)FF * NN * sizeof(float));
    int*    pcnt  = (int*)  (ws + 2 * PACK_BYTES + (size_t)FF * NN * sizeof(float)
                                + (size_t)FF * NT * sizeof(float));

    // 16384 (f,n) waves, 8 waves per block
    cosim_prep_pack<<<(FF * NN) / 8, 256, 0, stream>>>(out_matrix, Apack, Bpack, diag);
    // one workgroup per (f, i_tile)
    cosim_hinge_main<<<FF * NT, 256, 0, stream>>>(Apack, Bpack, diag, label,
                                                  margin, ploss, pcnt);
    cosim_finish<<<1, 256, 0, stream>>>(ploss, pcnt, margin, out);
}
